// GraphQNet_11751030522409
// MI455X (gfx1250) — matrix operand order repaired and checked by reading, not compiled, not run on hardware
//
#include <hip/hip_runtime.h>

#define HDIM   256
#define NBD    16
#define EID    8
#define LAYERS 6
#define NNODES 20000
#define NEDGES 320000
#define NGRAPH 20
#define PGRAPH 1000

typedef __bf16 bf16;
typedef __attribute__((ext_vector_type(8)))  bf16  v8bf;
typedef __attribute__((ext_vector_type(16))) bf16  v16bf;
typedef __attribute__((ext_vector_type(8)))  float v8f;

__device__ __forceinline__ float siluf(float x) { return x / (1.0f + __expf(-x)); }

__device__ __forceinline__ v16bf combine16(v8bf lo, v8bf hi) {
  v16bf f;
#pragma unroll
  for (int i = 0; i < 8; ++i) { f[i] = lo[i]; f[8 + i] = hi[i]; }
  return f;
}

// ---- Generic K=256, Ncol=256 WMMA GEMM with fused epilogues ----
// block = 256 threads = 8 waves; wave w computes rows [blk*128+w*16, +16) x 64 cols.
// B (the 64-col slice of pre-transposed W^T) is staged once per block into LDS in
// exact WMMA-fragment order: [kstep 0..7][tile 0..3][lane 0..31][16 bf16 = 32B].
// Both A (global) and B (LDS) fragments are register double-buffered across
// k-steps so VMEM and DS latency hide under the 4-independent-accumulator WMMA
// chain (waits become partial: dscnt <= outstanding-next instead of 0).
// mode 0: C = A@W + b              -> bf16 (in-place on A races: the 4 column blocks all read the same rows)
// mode 1: C = silu(A@W + b)        -> bf16
// mode 2: msg = relu(A@W + b + h[src]); atomicAdd(agg[dst], msg)   (GINE edge step)
// mode 3: C = A@W + b              -> f32
// mode 4: C = tanh(A@W + b)        -> f32
__global__ __launch_bounds__(256)
void gemm256_wmma(const bf16* __restrict__ A, const bf16* __restrict__ WT,
                  const float* __restrict__ bias, int M, int mode,
                  float* __restrict__ Cf, bf16* __restrict__ Cbf,
                  const float* __restrict__ hgat,
                  const int* __restrict__ src, const int* __restrict__ dst,
                  float* __restrict__ agg) {
  __shared__ bf16 bsm[8 * 4 * 32 * 16];  // 32 KB

  const int tid  = threadIdx.x;
  const int lane = tid & 31;
  const int wave = tid >> 5;
  const int row0 = blockIdx.x * 128 + wave * 16;
  const int col0 = blockIdx.y * 64;

  // ---- stage B slice into LDS, permuted to fragment order ----
  // chunk c = (s, t, l): WT row n = col0 + t*16 + (l&15); half 0 holds K = s*32 + (l>>4)*8 + 0..7,
  // half 1 holds K = s*32 + 16 + (l>>4)*8 + 0..7.  u indexes 16-byte halves of chunks.
#pragma unroll
  for (int u = tid; u < 2048; u += 256) {
    const int c = u >> 1, half = u & 1;
    const int s = c >> 7, t = (c >> 5) & 3, l = c & 31;
    const int n = col0 + t * 16 + (l & 15);
    const int gg = l >> 4;
    const bf16* gsrc = WT + (size_t)n * HDIM + s * 32 + gg * 8 + half * 16;
    *(v8bf*)(bsm + (size_t)c * 16 + half * 8) = *(const v8bf*)gsrc;
  }
  __syncthreads();

  int m_a = row0 + (lane & 15);
  if (m_a >= M) m_a = M - 1;
  const int ga = lane >> 4;
  // A fragment for k-step s: lo = Ap + s*32, hi = Ap + s*32 + 16
  const bf16* Ap = A + (size_t)m_a * HDIM + ga * 8;
  // B fragment base for this lane; step s, tile t at byte offset ((s*4+t)*32+lane)*32
  const bf16* Bp = bsm + (size_t)lane * 16;

  v8f acc[4];
#pragma unroll
  for (int t = 0; t < 4; ++t)
#pragma unroll
    for (int i = 0; i < 8; ++i) acc[t][i] = 0.0f;

  // prime step 0 buffers
  v8bf alo[2], ahi[2];
  v8bf blo[2][4], bhi[2][4];
  alo[0] = *(const v8bf*)(Ap);
  ahi[0] = *(const v8bf*)(Ap + 16);
#pragma unroll
  for (int t = 0; t < 4; ++t) {
    const bf16* bp = Bp + (size_t)t * 32 * 16;
    blo[0][t] = *(const v8bf*)bp;
    bhi[0][t] = *(const v8bf*)(bp + 8);
  }

#pragma unroll
  for (int s = 0; s < 8; ++s) {
    const int cur = s & 1, nxt = cur ^ 1;
    if (s < 7) {  // issue next step's A (global) and B (LDS) loads up-front
      alo[nxt] = *(const v8bf*)(Ap + (s + 1) * 32);
      ahi[nxt] = *(const v8bf*)(Ap + (s + 1) * 32 + 16);
#pragma unroll
      for (int t = 0; t < 4; ++t) {
        const bf16* bp = Bp + (size_t)((s + 1) * 4 + t) * 32 * 16;
        blo[nxt][t] = *(const v8bf*)bp;
        bhi[nxt][t] = *(const v8bf*)(bp + 8);
      }
    }
    const v16bf a = combine16(alo[cur], ahi[cur]);
#pragma unroll
    for (int t = 0; t < 4; ++t) {
      const v16bf b = combine16(blo[cur][t], bhi[cur][t]);
      acc[t] = __builtin_amdgcn_wmma_f32_16x16x32_bf16(
          false, a, false, b, (short)0, acc[t], false, false);
    }
  }

  const int g = lane >> 4;
  const int nloc = lane & 15;
#pragma unroll
  for (int r = 0; r < 8; ++r) {
    const int m = row0 + g * 8 + r;
    if (m >= M) continue;
    int s = 0, d = 0;
    if (mode == 2) { s = src[m]; d = dst[m]; }
#pragma unroll
    for (int t = 0; t < 4; ++t) {
      const int n = col0 + t * 16 + nloc;
      float v = acc[t][r] + bias[n];
      const size_t idx = (size_t)m * HDIM + n;
      if (mode == 0) {
        Cbf[idx] = (bf16)v;
      } else if (mode == 1) {
        Cbf[idx] = (bf16)siluf(v);
      } else if (mode == 2) {
        float msg = v + hgat[(size_t)s * HDIM + n];
        msg = msg > 0.0f ? msg : 0.0f;
        atomicAdd(&agg[(size_t)d * HDIM + n], msg);
      } else if (mode == 3) {
        Cf[idx] = v;
      } else {
        Cf[idx] = tanhf(v);
      }
    }
  }
}

// ---- prep: transpose 256x256 f32 weight to bf16 W^T ----
__global__ void transpose_bf16_kernel(const float* __restrict__ W, bf16* __restrict__ WT) {
  int idx = blockIdx.x * blockDim.x + threadIdx.x;
  if (idx >= HDIM * HDIM) return;
  int k = idx >> 8, n = idx & 255;
  WT[n * HDIM + k] = (bf16)W[idx];
}

// ---- edge MLP layer 1: t = relu(edge_attr @ ew1 + eb1), K=8 ----
__global__ void edge_mlp1_kernel(const float* __restrict__ ea, const float* __restrict__ ew1,
                                 const float* __restrict__ eb1, bf16* __restrict__ out) {
  size_t idx = (size_t)blockIdx.x * blockDim.x + threadIdx.x;
  if (idx >= (size_t)NEDGES * HDIM) return;
  int e = (int)(idx >> 8), c = (int)(idx & 255);
  float acc = eb1[c];
#pragma unroll
  for (int k = 0; k < EID; ++k) acc += ea[(size_t)e * EID + k] * ew1[k * HDIM + c];
  out[idx] = (bf16)(acc > 0.0f ? acc : 0.0f);
}

// ---- input proj: h = silu([x, c] @ in_w + in_b), K=17 ----
// node_gid*PG + node_lid == node index, so c_scalar[n] == c_flat[n].
__global__ void node_input_kernel(const float* __restrict__ x, const float* __restrict__ cfl,
                                  const float* __restrict__ in_w, const float* __restrict__ in_b,
                                  float* __restrict__ h, bf16* __restrict__ hbf) {
  size_t idx = (size_t)blockIdx.x * blockDim.x + threadIdx.x;
  if (idx >= (size_t)NNODES * HDIM) return;
  int n = (int)(idx >> 8), c = (int)(idx & 255);
  float acc = in_b[c];
#pragma unroll
  for (int k = 0; k < NBD; ++k) acc += x[(size_t)n * NBD + k] * in_w[k * HDIM + c];
  acc += cfl[n] * in_w[NBD * HDIM + c];
  float s = siluf(acc);
  h[idx] = s;
  hbf[idx] = (bf16)s;
}

// ---- z = bf16(h + agg) ----
__global__ void add_bf16_kernel(const float* __restrict__ h, const float* __restrict__ agg,
                                bf16* __restrict__ zbf) {
  size_t idx = (size_t)blockIdx.x * blockDim.x + threadIdx.x;
  if (idx >= (size_t)NNODES * HDIM) return;
  zbf[idx] = (bf16)(h[idx] + agg[idx]);
}

// ---- h += silu(layernorm(hn)*g + b); refresh bf16 mirror ----
__global__ __launch_bounds__(256)
void ln_silu_residual_kernel(const float* __restrict__ hn, const float* __restrict__ g,
                             const float* __restrict__ b, float* __restrict__ h,
                             bf16* __restrict__ hbf) {
  __shared__ float red[256];
  const int row = blockIdx.x, c = threadIdx.x;
  const size_t idx = (size_t)row * HDIM + c;
  float v = hn[idx];
  red[c] = v; __syncthreads();
  for (int s = 128; s > 0; s >>= 1) { if (c < s) red[c] += red[c + s]; __syncthreads(); }
  float mean = red[0] * (1.0f / HDIM); __syncthreads();
  float dv = v - mean;
  red[c] = dv * dv; __syncthreads();
  for (int s = 128; s > 0; s >>= 1) { if (c < s) red[c] += red[c + s]; __syncthreads(); }
  float var = red[0] * (1.0f / HDIM);
  float y = dv * rsqrtf(var + 1e-5f) * g[c] + b[c];
  float nh = h[idx] + siluf(y);
  h[idx] = nh;
  hbf[idx] = (bf16)nh;
}

// ---- scores[n] = dot(tanh_row, pa_w2) + pa_b2 ----
__global__ __launch_bounds__(256)
void attn_score_kernel(const float* __restrict__ a1, const float* __restrict__ w2,
                       const float* __restrict__ b2, float* __restrict__ scores) {
  __shared__ float red[256];
  const int row = blockIdx.x, c = threadIdx.x;
  red[c] = a1[(size_t)row * HDIM + c] * w2[c];
  __syncthreads();
  for (int s = 128; s > 0; s >>= 1) { if (c < s) red[c] += red[c + s]; __syncthreads(); }
  if (c == 0) scores[row] = red[0] + b2[0];
}

// ---- global softmax stats over all nodes ----
__global__ __launch_bounds__(1024)
void softmax_stats_kernel(const float* __restrict__ scores, float* __restrict__ stats) {
  __shared__ float red[1024];
  const int t = threadIdx.x;
  float mx = -1e30f;
  for (int i = t; i < NNODES; i += 1024) mx = fmaxf(mx, scores[i]);
  red[t] = mx; __syncthreads();
  for (int s = 512; s > 0; s >>= 1) { if (t < s) red[t] = fmaxf(red[t], red[t + s]); __syncthreads(); }
  float gmax = red[0]; __syncthreads();
  float sm = 0.0f;
  for (int i = t; i < NNODES; i += 1024) sm += __expf(scores[i] - gmax);
  red[t] = sm; __syncthreads();
  for (int s = 512; s > 0; s >>= 1) { if (t < s) red[t] += red[t + s]; __syncthreads(); }
  if (t == 0) { stats[0] = gmax; stats[1] = red[0]; }
}

// ---- attn = exp(s-max)/total; segsum per contiguous graph ----
__global__ __launch_bounds__(1024)
void graph_attn_kernel(const float* __restrict__ scores, const float* __restrict__ stats,
                       float* __restrict__ attn, float* __restrict__ segsum) {
  __shared__ float red[1024];
  const int b = blockIdx.x, t = threadIdx.x;
  const float gmax = stats[0], tot = stats[1];
  float w = 0.0f;
  if (t < PGRAPH) {
    int n = b * PGRAPH + t;
    w = __expf(scores[n] - gmax) / tot;
    attn[n] = w;
  }
  red[t] = w; __syncthreads();
  for (int s = 512; s > 0; s >>= 1) { if (t < s) red[t] += red[t + s]; __syncthreads(); }
  if (t == 0) segsum[b] = red[0];
}

// ---- pooled[b, c] = sum_n h[n,c] * attn[n]/(segsum[b]+1e-8) ----
__global__ __launch_bounds__(256)
void pool_kernel(const float* __restrict__ h, const float* __restrict__ attn,
                 const float* __restrict__ segsum, float* __restrict__ pooled) {
  const int b = blockIdx.x, c = threadIdx.x;
  const float denom = segsum[b] + 1e-8f;
  float acc = 0.0f;
  for (int i = 0; i < PGRAPH; ++i) {
    int n = b * PGRAPH + i;
    acc += h[(size_t)n * HDIM + c] * (attn[n] / denom);
  }
  pooled[b * HDIM + c] = acc;
}

// ---- head: q = silu(pooled @ hw1 + hb1) @ hw2 + hb2 ----
__global__ __launch_bounds__(256)
void head_kernel(const float* __restrict__ pooled, const float* __restrict__ hw1,
                 const float* __restrict__ hb1, const float* __restrict__ hw2,
                 const float* __restrict__ hb2, float* __restrict__ out) {
  __shared__ float red[256];
  __shared__ float prow[256];
  const int b = blockIdx.x, c = threadIdx.x;
  prow[c] = pooled[b * HDIM + c];
  __syncthreads();
  float acc = hb1[c];
  for (int k = 0; k < HDIM; ++k) acc += prow[k] * hw1[k * HDIM + c];
  red[c] = siluf(acc) * hw2[c];
  __syncthreads();
  for (int s = 128; s > 0; s >>= 1) { if (c < s) red[c] += red[c + s]; __syncthreads(); }
  if (c == 0) out[b] = red[0] + hb2[0];
}

extern "C" void kernel_launch(void* const* d_in, const int* in_sizes, int n_in,
                              void* d_out, int out_size, void* d_ws, size_t ws_size,
                              hipStream_t stream) {
  (void)in_sizes; (void)n_in; (void)out_size; (void)ws_size;
  const float* x    = (const float*)d_in[0];
  const float* ea   = (const float*)d_in[1];
  const float* cfl  = (const float*)d_in[2];
  const int*   eidx = (const int*)d_in[3];
  const float* ew1  = (const float*)d_in[6];
  const float* eb1  = (const float*)d_in[7];
  const float* ew2  = (const float*)d_in[8];
  const float* eb2  = (const float*)d_in[9];
  const float* in_w = (const float*)d_in[10];
  const float* in_b = (const float*)d_in[11];
  const float* lin_w= (const float*)d_in[12];
  const float* lin_b= (const float*)d_in[13];
  const float* cw1  = (const float*)d_in[14];
  const float* cb1  = (const float*)d_in[15];
  const float* cw2  = (const float*)d_in[16];
  const float* cb2  = (const float*)d_in[17];
  const float* ln_g = (const float*)d_in[18];
  const float* ln_b = (const float*)d_in[19];
  const float* paw1 = (const float*)d_in[20];
  const float* pab1 = (const float*)d_in[21];
  const float* paw2 = (const float*)d_in[22];
  const float* pab2 = (const float*)d_in[23];
  const float* hw1  = (const float*)d_in[24];
  const float* hb1  = (const float*)d_in[25];
  const float* hw2  = (const float*)d_in[26];
  const float* hb2  = (const float*)d_in[27];
  const int* src = eidx;
  const int* dst = eidx + NEDGES;
  float* out = (float*)d_out;

  char* ws = (char*)d_ws;
  size_t off = 0;
  auto take = [&](size_t bytes) -> char* {
    char* p = ws + off;
    off = (off + bytes + 255) & ~(size_t)255;
    return p;
  };
  const size_t NH = (size_t)NNODES * HDIM;
  const size_t EH = (size_t)NEDGES * HDIM;
  const size_t HH = (size_t)HDIM * HDIM;

  bf16*  ebf    = (bf16*)take(EH * sizeof(bf16));
  float* h      = (float*)take(NH * sizeof(float));
  bf16*  hbf    = (bf16*)take(NH * sizeof(bf16));
  float* agg    = (float*)take(NH * sizeof(float));
  bf16*  zbf    = (bf16*)take(NH * sizeof(bf16));
  float* hn     = (float*)take(NH * sizeof(float));   // also a1 for attention
  bf16*  t2bf   = (bf16*)take(NH * sizeof(bf16));
  float* scores = (float*)take(NNODES * sizeof(float));
  float* attn   = (float*)take(NNODES * sizeof(float));
  float* stats  = (float*)take(2 * sizeof(float));
  float* segsum = (float*)take(NGRAPH * sizeof(float));
  float* pooled = (float*)take((size_t)NGRAPH * HDIM * sizeof(float));
  bf16*  ew2T   = (bf16*)take(HH * sizeof(bf16));
  bf16*  linT   = (bf16*)take(LAYERS * HH * sizeof(bf16));
  bf16*  cw1T   = (bf16*)take(LAYERS * HH * sizeof(bf16));
  bf16*  cw2T   = (bf16*)take(LAYERS * HH * sizeof(bf16));
  bf16*  paw1T  = (bf16*)take(HH * sizeof(bf16));

  // --- weight prep: 20 transposes of 256x256 f32 -> bf16 W^T ---
  const int tB = (int)(HH / 256);
  transpose_bf16_kernel<<<tB, 256, 0, stream>>>(ew2, ew2T);
  transpose_bf16_kernel<<<tB, 256, 0, stream>>>(paw1, paw1T);
  for (int l = 0; l < LAYERS; ++l) {
    transpose_bf16_kernel<<<tB, 256, 0, stream>>>(lin_w + l * HH, linT + l * HH);
    transpose_bf16_kernel<<<tB, 256, 0, stream>>>(cw1 + l * HH, cw1T + l * HH);
    transpose_bf16_kernel<<<tB, 256, 0, stream>>>(cw2 + l * HH, cw2T + l * HH);
  }

  // --- edge MLP: t = relu(ea@ew1+eb1); e = t@ew2+eb2 (WMMA, in-place bf16) ---
  edge_mlp1_kernel<<<(int)(EH / 256), 256, 0, stream>>>(ea, ew1, eb1, ebf);
  dim3 egrid(NEDGES / 128, 4);
  gemm256_wmma<<<egrid, 256, 0, stream>>>(ebf, ew2T, eb2, NEDGES, 0,
                                          nullptr, ebf, nullptr, nullptr, nullptr, nullptr);

  // --- input projection ---
  node_input_kernel<<<(int)(NH / 256), 256, 0, stream>>>(x, cfl, in_w, in_b, h, hbf);

  dim3 ngrid((NNODES + 127) / 128, 4);
  for (int l = 0; l < LAYERS; ++l) {
    hipMemsetAsync(agg, 0, NH * sizeof(float), stream);
    // fused: g = e@lin_w[l]+lin_b[l]; msg = relu(g + h[src]); agg[dst] += msg
    gemm256_wmma<<<egrid, 256, 0, stream>>>(ebf, linT + l * HH, lin_b + l * HDIM, NEDGES, 2,
                                            nullptr, nullptr, h, src, dst, agg);
    // z = h + agg
    add_bf16_kernel<<<(int)(NH / 256), 256, 0, stream>>>(h, agg, zbf);
    // t2 = silu(z@cw1+cb1)
    gemm256_wmma<<<ngrid, 256, 0, stream>>>(zbf, cw1T + l * HH, cb1 + l * HDIM, NNODES, 1,
                                            nullptr, t2bf, nullptr, nullptr, nullptr, nullptr);
    // hn = t2@cw2+cb2
    gemm256_wmma<<<ngrid, 256, 0, stream>>>(t2bf, cw2T + l * HH, cb2 + l * HDIM, NNODES, 3,
                                            hn, nullptr, nullptr, nullptr, nullptr, nullptr);
    // h += silu(LN(hn))
    ln_silu_residual_kernel<<<NNODES, 256, 0, stream>>>(hn, ln_g + l * HDIM, ln_b + l * HDIM, h, hbf);
  }

  // --- attention pooling ---
  gemm256_wmma<<<ngrid, 256, 0, stream>>>(hbf, paw1T, pab1, NNODES, 4,
                                          hn, nullptr, nullptr, nullptr, nullptr, nullptr);
  attn_score_kernel<<<NNODES, 256, 0, stream>>>(hn, paw2, pab2, scores);
  softmax_stats_kernel<<<1, 1024, 0, stream>>>(scores, stats);
  graph_attn_kernel<<<NGRAPH, 1024, 0, stream>>>(scores, stats, attn, segsum);
  pool_kernel<<<NGRAPH, 256, 0, stream>>>(h, attn, segsum, pooled);
  head_kernel<<<NGRAPH, 256, 0, stream>>>(pooled, hw1, hb1, hw2, hb2, out);
}
